// ACTSmooth_49065706389540
// MI455X (gfx1250) — matrix operand order repaired and checked by reading, not compiled, not run on hardware
//
#include <hip/hip_runtime.h>

typedef __attribute__((ext_vector_type(16))) _Float16 v16h;
typedef __attribute__((ext_vector_type(8)))  _Float16 v8h;
typedef __attribute__((ext_vector_type(8)))  float    v8f;

#define BSZ   8
#define NH    8
#define HD    64
#define DM    512
#define FFND  3200
#define NSH   604
#define NBT   5
#define CHK   100
#define SENC  629
#define SDEC  500
#define ME    (SENC*BSZ)   /* 5032 */
#define MD    (SDEC*BSZ)   /* 4000 */

__device__ __forceinline__ int imin(int a, int b) { return a < b ? a : b; }

// ---------------- WMMA fragment loaders (per CDNA5 ISA 7.12.2), pure f16 ----------------
// A-matrix 16x32 f16: lane holds row M=lane&15; halves 0..7 -> K=kb..kb+7,
// halves 8..15 -> K=kb+16..kb+23, kb = 8*(lane>>4).
__device__ __forceinline__ v16h load_a(const _Float16* __restrict__ p, int ld, int row, int k0, int lane) {
    const _Float16* r = p + (size_t)row * ld + k0 + ((lane >> 4) << 3);
    v8h x0 = *(const v8h*)(r);
    v8h x1 = *(const v8h*)(r + 16);
    v16h f;
#pragma unroll
    for (int i = 0; i < 8; ++i) { f[i] = x0[i]; f[8 + i] = x1[i]; }
    return f;
}
// B-matrix 32x16 f16 (K x N), sourced from row-major W[N,K] (column-major B):
// lane holds column N=lane&15; halves 0..7 -> K=kb..kb+7, halves 8..15 -> K=kb+16..kb+23, kb = 8*(lane>>4).
__device__ __forceinline__ v16h load_b(const _Float16* __restrict__ p, int ld, int row, int k0, int lane) {
    const _Float16* r = p + (size_t)row * ld + k0 + ((lane >> 4) << 3);
    v8h x0 = *(const v8h*)(r);
    v8h x1 = *(const v8h*)(r + 16);
    v16h f;
#pragma unroll
    for (int i = 0; i < 8; ++i) { f[i] = x0[i]; f[8 + i] = x1[i]; }
    return f;
}

// ---------------- batched GEMM: C[bz] = act(A[bz] @ B[bz]^T + bias) ----------------
// A: [M,K] f16, B: [N,K] f16 K-major, C: fp32 and/or f16 (either may be null).
// Block = 256 thr = 8 waves arranged 2(M) x 4(N); wave tile 32x64 -> 8 wmma / K-step.
template <int ACT>
__global__ __launch_bounds__(256) void gemm_wmma(
    const _Float16* __restrict__ A, int lda, long long sA,
    const _Float16* __restrict__ B, int ldb, long long sB,
    const float* __restrict__ bias,
    float* __restrict__ C, _Float16* __restrict__ Ch, int ldc, long long sC,
    int M, int N, int K)
{
    A += (size_t)blockIdx.z * (size_t)sA;
    B += (size_t)blockIdx.z * (size_t)sB;
    if (C)  C  += (size_t)blockIdx.z * (size_t)sC;
    if (Ch) Ch += (size_t)blockIdx.z * (size_t)sC;
    const int lane = threadIdx.x & 31;
    const int wave = threadIdx.x >> 5;
    const int m0 = blockIdx.y * 64 + (wave & 1) * 32;
    const int n0 = blockIdx.x * 256 + (wave >> 1) * 64;
    if (m0 >= M || n0 >= N) return;  // wave-uniform; EXEC stays all-ones for WMMA

    const int r   = lane & 15;
    const int rA0 = imin(m0 + r, M - 1);
    const int rA1 = imin(m0 + 16 + r, M - 1);
    int cB[4];
#pragma unroll
    for (int j = 0; j < 4; ++j) cB[j] = imin(n0 + 16 * j + r, N - 1);

    v8f acc[2][4];
#pragma unroll
    for (int i = 0; i < 2; ++i)
#pragma unroll
        for (int j = 0; j < 4; ++j)
#pragma unroll
            for (int t = 0; t < 8; ++t) acc[i][j][t] = 0.0f;

    for (int k0 = 0; k0 < K; k0 += 32) {
        v16h a0 = load_a(A, lda, rA0, k0, lane);
        v16h a1 = load_a(A, lda, rA1, k0, lane);
        v16h b0 = load_b(B, ldb, cB[0], k0, lane);
        v16h b1 = load_b(B, ldb, cB[1], k0, lane);
        v16h b2 = load_b(B, ldb, cB[2], k0, lane);
        v16h b3 = load_b(B, ldb, cB[3], k0, lane);
        acc[0][0] = __builtin_amdgcn_wmma_f32_16x16x32_f16(false, a0, false, b0, (short)0, acc[0][0], false, false);
        acc[0][1] = __builtin_amdgcn_wmma_f32_16x16x32_f16(false, a0, false, b1, (short)0, acc[0][1], false, false);
        acc[0][2] = __builtin_amdgcn_wmma_f32_16x16x32_f16(false, a0, false, b2, (short)0, acc[0][2], false, false);
        acc[0][3] = __builtin_amdgcn_wmma_f32_16x16x32_f16(false, a0, false, b3, (short)0, acc[0][3], false, false);
        acc[1][0] = __builtin_amdgcn_wmma_f32_16x16x32_f16(false, a1, false, b0, (short)0, acc[1][0], false, false);
        acc[1][1] = __builtin_amdgcn_wmma_f32_16x16x32_f16(false, a1, false, b1, (short)0, acc[1][1], false, false);
        acc[1][2] = __builtin_amdgcn_wmma_f32_16x16x32_f16(false, a1, false, b2, (short)0, acc[1][2], false, false);
        acc[1][3] = __builtin_amdgcn_wmma_f32_16x16x32_f16(false, a1, false, b3, (short)0, acc[1][3], false, false);
    }

    // C/D layout: lane 0-15 -> N=lane, VGPR t -> M=t (lanes<16) / t+8 (lanes>=16)
    const int mh = (lane >> 4) * 8;
#pragma unroll
    for (int j = 0; j < 4; ++j) {
        const int n = n0 + 16 * j + r;
        const bool nok = n < N;
        const float bv = (bias != nullptr && nok) ? bias[n] : 0.0f;
#pragma unroll
        for (int i = 0; i < 2; ++i) {
#pragma unroll
            for (int t = 0; t < 8; ++t) {
                const int m = m0 + 16 * i + mh + t;
                float v = acc[i][j][t] + bv;
                if (ACT == 1) v = fmaxf(v, 0.0f);
                if (nok && m < M) {
                    if (C)  C[(size_t)m * ldc + n] = v;
                    if (Ch) Ch[(size_t)m * ldc + n] = (_Float16)v;
                }
            }
        }
    }
}

// ---------------- reductions ----------------
__device__ __forceinline__ float block_reduce_sum(float v, float* sred) {
#pragma unroll
    for (int o = 16; o > 0; o >>= 1) v += __shfl_xor(v, o, 32);
    const int lane = threadIdx.x & 31, w = threadIdx.x >> 5;
    __syncthreads();
    if (lane == 0) sred[w] = v;
    __syncthreads();
    float rr = 0.0f;
    const int nw = blockDim.x >> 5;
    for (int j = 0; j < nw; ++j) rr += sred[j];
    return rr;
}
__device__ __forceinline__ float block_reduce_max(float v, float* sred) {
#pragma unroll
    for (int o = 16; o > 0; o >>= 1) v = fmaxf(v, __shfl_xor(v, o, 32));
    const int lane = threadIdx.x & 31, w = threadIdx.x >> 5;
    __syncthreads();
    if (lane == 0) sred[w] = v;
    __syncthreads();
    float rr = sred[0];
    const int nw = blockDim.x >> 5;
    for (int j = 1; j < nw; ++j) rr = fmaxf(rr, sred[j]);
    return rr;
}

// ---------------- analytic VLASH block masks ----------------
// kind 0: encoder self, kind 1: decoder self, kind 2: decoder cross
__device__ __forceinline__ bool attn_allowed(int kind, int q, int k) {
    if (kind == 0) return (k < NSH) || ((q >= NSH) && ((q - NSH) / NBT == (k - NSH) / NBT));
    if (kind == 1) return (q / CHK) == (k / CHK);
    return (k < NSH) || ((k - NSH) / NBT == (q / CHK));
}

// ---------------- masked softmax: fp32 scores -> f16 probs (zero-padded to SkPad) ----
__global__ __launch_bounds__(256) void softmax_mask_kernel(
    const float* __restrict__ scores, _Float16* __restrict__ probs,
    int Sq, int Sk, int SkPad, int kind)
{
    __shared__ float sred[8];
    const int q = blockIdx.x, bh = blockIdx.y;
    const float* row = scores + ((size_t)bh * Sq + q) * Sk;
    _Float16* prow = probs + ((size_t)bh * Sq + q) * SkPad;
    float mx = -3.0e38f;
    for (int k = threadIdx.x; k < Sk; k += 256)
        if (attn_allowed(kind, q, k)) mx = fmaxf(mx, row[k]);
    mx = block_reduce_max(mx, sred);
    float s = 0.0f;
    for (int k = threadIdx.x; k < Sk; k += 256) {
        const float e = attn_allowed(kind, q, k) ? __expf(row[k] - mx) : 0.0f;
        prow[k] = (_Float16)e;
        s += e;
    }
    s = block_reduce_sum(s, sred);
    const float inv = 1.0f / s;
    for (int k = threadIdx.x; k < Sk; k += 256) prow[k] = (_Float16)((float)prow[k] * inv);
    for (int k = Sk + threadIdx.x; k < SkPad; k += 256) prow[k] = (_Float16)0.0f;
}

// ---------------- LayerNorm (+ optional residual), D = 512, one token / block ----
// Writes fp32 (for later residuals) and optionally an f16 mirror (GEMM operand).
__global__ __launch_bounds__(256) void ln_kernel(
    const float* __restrict__ x, const float* __restrict__ res,
    const float* __restrict__ g, const float* __restrict__ b,
    float* __restrict__ out, _Float16* __restrict__ outh)
{
    __shared__ float sred[8];
    const size_t base = (size_t)blockIdx.x * DM;
    const int i0 = threadIdx.x, i1 = threadIdx.x + 256;
    float v0 = x[base + i0] + (res ? res[base + i0] : 0.0f);
    float v1 = x[base + i1] + (res ? res[base + i1] : 0.0f);
    const float mu = block_reduce_sum(v0 + v1, sred) * (1.0f / DM);
    const float d0 = v0 - mu, d1 = v1 - mu;
    const float var = block_reduce_sum(d0 * d0 + d1 * d1, sred) * (1.0f / DM);
    const float inv = rsqrtf(var + 1e-5f);
    const float o0 = d0 * inv * g[i0] + b[i0];
    const float o1 = d1 * inv * g[i1] + b[i1];
    out[base + i0] = o0;
    out[base + i1] = o1;
    if (outh) {
        outh[base + i0] = (_Float16)o0;
        outh[base + i1] = (_Float16)o1;
    }
}

// ---------------- elementwise helpers ----------------
// x + pos is only ever a GEMM operand -> emit f16 directly
__global__ __launch_bounds__(256) void addpos_kernel(
    const float* __restrict__ x, const float* __restrict__ pos, _Float16* __restrict__ out, int S)
{
    const int n = S * BSZ * DM;
    for (int i = blockIdx.x * blockDim.x + threadIdx.x; i < n; i += gridDim.x * blockDim.x) {
        const int d = i & (DM - 1);
        const int s = i / (BSZ * DM);
        out[i] = (_Float16)(x[i] + pos[s * DM + d]);
    }
}

// fp32 -> f16 (weights, once per launch)
__global__ __launch_bounds__(256) void cvt_f16_kernel(
    const float* __restrict__ src, _Float16* __restrict__ dst, int n)
{
    for (int i = blockIdx.x * blockDim.x + threadIdx.x; i < n; i += gridDim.x * blockDim.x)
        dst[i] = (_Float16)src[i];
}

// copy fp32 (residual base) + f16 mirror (GEMM operand)
__global__ __launch_bounds__(256) void copy_cvt_kernel(
    const float* __restrict__ src, float* __restrict__ dst, _Float16* __restrict__ dsth, int n)
{
    for (int i = blockIdx.x * blockDim.x + threadIdx.x; i < n; i += gridDim.x * blockDim.x) {
        const float v = src[i];
        dst[i] = v;
        dsth[i] = (_Float16)v;
    }
}

// src fp32 [S*B, ld] (token t = s*B+b), column block co + h*64 + d -> f16 [bh][s][d]
__global__ __launch_bounds__(256) void pack_head_kernel(
    const float* __restrict__ src, int ld, int co, _Float16* __restrict__ dst, int S, float scale)
{
    const int n = BSZ * NH * S * HD;
    for (int i = blockIdx.x * blockDim.x + threadIdx.x; i < n; i += gridDim.x * blockDim.x) {
        const int d = i & 63;
        const int s = (i >> 6) % S;
        const int bh = i / (S * HD);
        const int b = bh >> 3, h = bh & 7;
        dst[i] = (_Float16)(src[(size_t)(s * BSZ + b) * ld + co + h * HD + d] * scale);
    }
}

// src fp32 [S*B, 512] -> Vt f16 [bh][d][s], zero-padded to SPad along s
__global__ __launch_bounds__(256) void pack_vt_kernel(
    const float* __restrict__ src, _Float16* __restrict__ dst, int S, int SPad)
{
    const int n = BSZ * NH * HD * SPad;
    for (int i = blockIdx.x * blockDim.x + threadIdx.x; i < n; i += gridDim.x * blockDim.x) {
        const int s = i % SPad;
        const int d = (i / SPad) & 63;
        const int bh = i / (SPad * HD);
        const int b = bh >> 3, h = bh & 7;
        dst[i] = (s < S) ? (_Float16)src[(size_t)(s * BSZ + b) * DM + h * HD + d] : (_Float16)0.0f;
    }
}

// attn fp32 [bh][s][d] -> ctx f16 [s*B+b][h*64+d] (only feeds out-proj GEMM)
__global__ __launch_bounds__(256) void unpack_ctx_kernel(
    const float* __restrict__ attn, _Float16* __restrict__ ctx, int S)
{
    const int n = S * BSZ * DM;
    for (int i = blockIdx.x * blockDim.x + threadIdx.x; i < n; i += gridDim.x * blockDim.x) {
        const int dd = i & (DM - 1);
        const int t = i >> 9;
        const int b = t & (BSZ - 1);
        const int s = t >> 3;
        const int h = dd >> 6, d = dd & 63;
        ctx[i] = (_Float16)attn[(((size_t)(b * NH + h) * S + s) << 6) + d];
    }
}

// =====================================================================
extern "C" void kernel_launch(void* const* d_in, const int* in_sizes, int n_in,
                              void* d_out, int out_size, void* d_ws, size_t ws_size,
                              hipStream_t stream)
{
    (void)in_sizes; (void)n_in; (void)out_size; (void)ws_size;
    const float* x_enc     = (const float*)d_in[0];
    const float* enc_pos   = (const float*)d_in[1];
    const float* x_dec     = (const float*)d_in[2];
    const float* dec_pos   = (const float*)d_in[3];
    const float* enc_in_w  = (const float*)d_in[4];
    const float* enc_in_b  = (const float*)d_in[5];
    const float* enc_out_w = (const float*)d_in[6];
    const float* enc_out_b = (const float*)d_in[7];
    const float* enc_l1_w  = (const float*)d_in[8];
    const float* enc_l1_b  = (const float*)d_in[9];
    const float* enc_l2_w  = (const float*)d_in[10];
    const float* enc_l2_b  = (const float*)d_in[11];
    const float* enc_n1_g  = (const float*)d_in[12];
    const float* enc_n1_b  = (const float*)d_in[13];
    const float* enc_n2_g  = (const float*)d_in[14];
    const float* enc_n2_b  = (const float*)d_in[15];
    const float* dec_sa_in_w  = (const float*)d_in[16];
    const float* dec_sa_in_b  = (const float*)d_in[17];
    const float* dec_sa_out_w = (const float*)d_in[18];
    const float* dec_sa_out_b = (const float*)d_in[19];
    const float* dec_ca_in_w  = (const float*)d_in[20];
    const float* dec_ca_in_b  = (const float*)d_in[21];
    const float* dec_ca_out_w = (const float*)d_in[22];
    const float* dec_ca_out_b = (const float*)d_in[23];
    const float* dec_l1_w  = (const float*)d_in[24];
    const float* dec_l1_b  = (const float*)d_in[25];
    const float* dec_l2_w  = (const float*)d_in[26];
    const float* dec_l2_b  = (const float*)d_in[27];
    const float* dec_n1_g  = (const float*)d_in[28];
    const float* dec_n1_b  = (const float*)d_in[29];
    const float* dec_n2_g  = (const float*)d_in[30];
    const float* dec_n2_b  = (const float*)d_in[31];
    const float* dec_n3_g  = (const float*)d_in[32];
    const float* dec_n3_b  = (const float*)d_in[33];
    const float* dec_norm_g = (const float*)d_in[34];
    const float* dec_norm_b = (const float*)d_in[35];

    // ---- workspace carve (256B aligned) ----
    char* wsb = (char*)d_ws;
    size_t off = 0;
    auto alloc = [&](size_t bytes) -> void* {
        off = (off + 255) & ~(size_t)255;
        void* p = wsb + off;
        off += bytes;
        return p;
    };
    // fp32 activations (residual bases / pre-softmax)
    float* encX  = (float*)alloc((size_t)ME * DM * 4);
    float* encY  = (float*)alloc((size_t)ME * DM * 4);
    float* tmpP  = (float*)alloc((size_t)ME * DM * 4);   // attn/ffn projection outputs (residual adds)
    float* decA  = (float*)alloc((size_t)MD * DM * 4);
    float* decB  = (float*)alloc((size_t)MD * DM * 4);
    float* qk    = (float*)alloc((size_t)ME * 1024 * 4);
    float* vb    = (float*)alloc((size_t)ME * DM * 4);
    float* crossq = (float*)alloc((size_t)MD * DM * 4);
    float* scores = (float*)alloc((size_t)BSZ * NH * SENC * SENC * 4);
    float* attnb = (float*)alloc((size_t)BSZ * NH * SENC * HD * 4);
    // f16 GEMM operands
    _Float16* encXh = (_Float16*)alloc((size_t)ME * DM * 2);
    _Float16* encYh = (_Float16*)alloc((size_t)ME * DM * 2);
    _Float16* tmpH  = (_Float16*)alloc((size_t)ME * DM * 2);  // (x+pos) f16
    _Float16* memH  = (_Float16*)alloc((size_t)ME * DM * 2);  // enc_out + enc_pos
    _Float16* decAh = (_Float16*)alloc((size_t)MD * DM * 2);
    _Float16* decBh = (_Float16*)alloc((size_t)MD * DM * 2);
    _Float16* ctxh  = (_Float16*)alloc((size_t)ME * DM * 2);
    _Float16* ffmh  = (_Float16*)alloc((size_t)ME * FFND * 2);
    _Float16* Qh = (_Float16*)alloc((size_t)BSZ * NH * SENC * HD * 2);
    _Float16* Kh = (_Float16*)alloc((size_t)BSZ * NH * SENC * HD * 2);
    _Float16* Vt = (_Float16*)alloc((size_t)BSZ * NH * HD * 640 * 2);
    _Float16* probs = (_Float16*)alloc((size_t)BSZ * NH * SENC * 640 * 2);
    // f16 weights (converted once per launch)
    _Float16* w_enc_in  = (_Float16*)alloc((size_t)4 * 1536 * DM * 2);
    _Float16* w_enc_out = (_Float16*)alloc((size_t)4 * DM * DM * 2);
    _Float16* w_enc_l1  = (_Float16*)alloc((size_t)4 * FFND * DM * 2);
    _Float16* w_enc_l2  = (_Float16*)alloc((size_t)4 * DM * FFND * 2);
    _Float16* w_sa_in   = (_Float16*)alloc((size_t)4 * 1536 * DM * 2);
    _Float16* w_sa_out  = (_Float16*)alloc((size_t)4 * DM * DM * 2);
    _Float16* w_ca_in   = (_Float16*)alloc((size_t)4 * 1536 * DM * 2);
    _Float16* w_ca_out  = (_Float16*)alloc((size_t)4 * DM * DM * 2);
    _Float16* w_dec_l1  = (_Float16*)alloc((size_t)4 * FFND * DM * 2);
    _Float16* w_dec_l2  = (_Float16*)alloc((size_t)4 * DM * FFND * 2);

    auto ewb = [](int n) { return dim3((unsigned)((n + 255) / 256)); };
    auto cvt = [&](const float* s, _Float16* d, int n) {
        cvt_f16_kernel<<<ewb(n), 256, 0, stream>>>(s, d, n);
    };

    // weights -> f16, once per launch
    cvt(enc_in_w,  w_enc_in,  4 * 1536 * DM);
    cvt(enc_out_w, w_enc_out, 4 * DM * DM);
    cvt(enc_l1_w,  w_enc_l1,  4 * FFND * DM);
    cvt(enc_l2_w,  w_enc_l2,  4 * DM * FFND);
    cvt(dec_sa_in_w,  w_sa_in,  4 * 1536 * DM);
    cvt(dec_sa_out_w, w_sa_out, 4 * DM * DM);
    cvt(dec_ca_in_w,  w_ca_in,  4 * 1536 * DM);
    cvt(dec_ca_out_w, w_ca_out, 4 * DM * DM);
    cvt(dec_l1_w, w_dec_l1, 4 * FFND * DM);
    cvt(dec_l2_w, w_dec_l2, 4 * DM * FFND);

    auto gemm_w = [&](const _Float16* A, int lda, const _Float16* Bw, int ldb, const float* bias,
                      float* C, _Float16* Ch, int ldc, int M, int N, int K, int act) {
        dim3 g((N + 255) / 256, (M + 63) / 64, 1);
        if (act)
            gemm_wmma<1><<<g, 256, 0, stream>>>(A, lda, 0LL, Bw, ldb, 0LL, bias, C, Ch, ldc, 0LL, M, N, K);
        else
            gemm_wmma<0><<<g, 256, 0, stream>>>(A, lda, 0LL, Bw, ldb, 0LL, bias, C, Ch, ldc, 0LL, M, N, K);
    };
    auto gemm_bh = [&](const _Float16* A, int lda, long long sA,
                       const _Float16* Bh, int ldb, long long sB,
                       float* C, int ldc, long long sC, int M, int N, int K) {
        dim3 g((N + 255) / 256, (M + 63) / 64, BSZ * NH);
        gemm_wmma<0><<<g, 256, 0, stream>>>(A, lda, sA, Bh, ldb, sB,
                                            (const float*)nullptr, C, (_Float16*)nullptr, ldc, sC, M, N, K);
    };

    // attention: packs -> QK^T -> masked softmax -> PV -> unpack(f16) -> out-proj
    auto attention = [&](const float* qsrc, int qld, int qco,
                         const float* ksrc, int kld, int kco,
                         const float* vsrc, int Sq, int Sk, int SkPad, int kind,
                         const _Float16* ow, const float* ob, float* projOut) {
        pack_head_kernel<<<ewb(BSZ * NH * Sq * HD), 256, 0, stream>>>(qsrc, qld, qco, Qh, Sq, 0.125f);
        pack_head_kernel<<<ewb(BSZ * NH * Sk * HD), 256, 0, stream>>>(ksrc, kld, kco, Kh, Sk, 1.0f);
        pack_vt_kernel<<<ewb(BSZ * NH * HD * SkPad), 256, 0, stream>>>(vsrc, Vt, Sk, SkPad);
        gemm_bh(Qh, HD, (long long)Sq * HD, Kh, HD, (long long)Sk * HD,
                scores, Sk, (long long)Sq * Sk, Sq, Sk, HD);
        softmax_mask_kernel<<<dim3(Sq, BSZ * NH), 256, 0, stream>>>(scores, probs, Sq, Sk, SkPad, kind);
        gemm_bh(probs, SkPad, (long long)Sq * SkPad, Vt, SkPad, (long long)HD * SkPad,
                attnb, HD, (long long)Sq * HD, Sq, HD, SkPad);
        unpack_ctx_kernel<<<ewb(Sq * BSZ * DM), 256, 0, stream>>>(attnb, ctxh, Sq);
        gemm_w(ctxh, DM, ow, DM, ob, projOut, nullptr, DM, Sq * BSZ, DM, DM, 0);
    };

    // ================= encoder =================
    copy_cvt_kernel<<<ewb(ME * DM), 256, 0, stream>>>(x_enc, encX, encXh, ME * DM);
    for (int L = 0; L < 4; ++L) {
        const _Float16* inw = w_enc_in + (size_t)L * 1536 * DM;
        const float* inb = enc_in_b + (size_t)L * 1536;
        const _Float16* ow = w_enc_out + (size_t)L * DM * DM;
        const float* ob = enc_out_b + (size_t)L * DM;
        const _Float16* l1w = w_enc_l1 + (size_t)L * FFND * DM;
        const float* l1b = enc_l1_b + (size_t)L * FFND;
        const _Float16* l2w = w_enc_l2 + (size_t)L * DM * FFND;
        const float* l2b = enc_l2_b + (size_t)L * DM;
        addpos_kernel<<<ewb(ME * DM), 256, 0, stream>>>(encX, enc_pos, tmpH, SENC);
        gemm_w(tmpH, DM, inw, DM, inb, qk, nullptr, 1024, ME, 1024, DM, 0);                       // Q|K from x+pos
        gemm_w(encXh, DM, inw + (size_t)1024 * DM, DM, inb + 1024, vb, nullptr, DM, ME, DM, DM, 0); // V from x
        attention(qk, 1024, 0, qk, 1024, 512, vb, SENC, SENC, 640, 0, ow, ob, tmpP);
        ln_kernel<<<ME, 256, 0, stream>>>(encX, tmpP, enc_n1_g + L * DM, enc_n1_b + L * DM, encY, encYh);
        gemm_w(encYh, DM, l1w, DM, l1b, nullptr, ffmh, FFND, ME, FFND, DM, 1);                    // FFN1+relu -> f16 only
        gemm_w(ffmh, FFND, l2w, FFND, l2b, tmpP, nullptr, DM, ME, DM, FFND, 0);                   // FFN2
        ln_kernel<<<ME, 256, 0, stream>>>(encY, tmpP, enc_n2_g + L * DM, enc_n2_b + L * DM, encX, encXh);
    }
    // cross-attn memory: enc_out + enc_pos (K path), f16
    addpos_kernel<<<ewb(ME * DM), 256, 0, stream>>>(encX, enc_pos, memH, SENC);

    // ================= decoder =================
    copy_cvt_kernel<<<ewb(MD * DM), 256, 0, stream>>>(x_dec, decA, decAh, MD * DM);
    float* cur = decA;      float* alt = decB;
    _Float16* curh = decAh; _Float16* alth = decBh;
    for (int L = 0; L < 4; ++L) {
        const _Float16* sain = w_sa_in + (size_t)L * 1536 * DM;
        const float* sainb = dec_sa_in_b + (size_t)L * 1536;
        const _Float16* saow = w_sa_out + (size_t)L * DM * DM;
        const float* saob = dec_sa_out_b + (size_t)L * DM;
        const _Float16* cain = w_ca_in + (size_t)L * 1536 * DM;
        const float* cainb = dec_ca_in_b + (size_t)L * 1536;
        const _Float16* caow = w_ca_out + (size_t)L * DM * DM;
        const float* caob = dec_ca_out_b + (size_t)L * DM;
        const _Float16* l1w = w_dec_l1 + (size_t)L * FFND * DM;
        const float* l1b = dec_l1_b + (size_t)L * FFND;
        const _Float16* l2w = w_dec_l2 + (size_t)L * DM * FFND;
        const float* l2b = dec_l2_b + (size_t)L * DM;

        // --- self-attention (block-diagonal per branch) ---
        addpos_kernel<<<ewb(MD * DM), 256, 0, stream>>>(cur, dec_pos, tmpH, SDEC);
        gemm_w(tmpH, DM, sain, DM, sainb, qk, nullptr, 1024, MD, 1024, DM, 0);
        gemm_w(curh, DM, sain + (size_t)1024 * DM, DM, sainb + 1024, vb, nullptr, DM, MD, DM, DM, 0);
        attention(qk, 1024, 0, qk, 1024, 512, vb, SDEC, SDEC, 512, 1, saow, saob, tmpP);
        ln_kernel<<<MD, 256, 0, stream>>>(cur, tmpP, dec_n1_g + L * DM, dec_n1_b + L * DM, alt, alth);

        // --- cross-attention (shared + own-branch encoder tokens) ---
        addpos_kernel<<<ewb(MD * DM), 256, 0, stream>>>(alt, dec_pos, tmpH, SDEC);
        gemm_w(tmpH, DM, cain, DM, cainb, crossq, nullptr, DM, MD, DM, DM, 0);                          // Q (dec)
        gemm_w(memH, DM, cain + (size_t)512 * DM, DM, cainb + 512, qk, nullptr, DM, ME, DM, DM, 0);     // K (enc mem)
        gemm_w(encXh, DM, cain + (size_t)1024 * DM, DM, cainb + 1024, vb, nullptr, DM, ME, DM, DM, 0);  // V (enc)
        attention(crossq, DM, 0, qk, DM, 0, vb, SDEC, SENC, 640, 2, caow, caob, tmpP);
        ln_kernel<<<MD, 256, 0, stream>>>(alt, tmpP, dec_n2_g + L * DM, dec_n2_b + L * DM, cur, curh);

        // --- FFN ---
        gemm_w(curh, DM, l1w, DM, l1b, nullptr, ffmh, FFND, MD, FFND, DM, 1);
        gemm_w(ffmh, FFND, l2w, FFND, l2b, tmpP, nullptr, DM, MD, DM, FFND, 0);
        ln_kernel<<<MD, 256, 0, stream>>>(cur, tmpP, dec_n3_g + L * DM, dec_n3_b + L * DM, alt, alth);

        { float* t = cur; cur = alt; alt = t; }
        { _Float16* t = curh; curh = alth; alth = t; }
    }
    // final norm straight into d_out (no f16 mirror needed)
    ln_kernel<<<MD, 256, 0, stream>>>(cur, (const float*)nullptr, dec_norm_g, dec_norm_b,
                                      (float*)d_out, (_Float16*)nullptr);
}
